// CongestionGAT_72808285602084
// MI455X (gfx1250) — compile-verified, matrix operand order checked by reading, not run on hardware
//
#include <hip/hip_runtime.h>
#include <hip/hip_bf16.h>

typedef __attribute__((ext_vector_type(16))) __bf16 v16bf;
typedef __attribute__((ext_vector_type(8)))  __bf16 v8bf;
typedef __attribute__((ext_vector_type(8)))  float  v8f;
typedef __attribute__((ext_vector_type(4)))  int    i4v;

#define HID 128
#define HEADS 4
#define DHEAD 32
#define IN_DIM 12
#define NEG_SLOPE 0.2f
#define BN_EPS 1e-5f

// ---------------------------------------------------------------------------
// async global->LDS staging (CDNA5 GLOBAL_LOAD_ASYNC_TO_LDS_B128, ASYNCcnt),
// with a synchronous fallback if the builtin is not available.
// Builtin signature (from hipcc diagnostic): param 1 = int4* in AS(1).
// ---------------------------------------------------------------------------
#if __has_builtin(__builtin_amdgcn_global_load_async_to_lds_b128)
#define HAVE_ASYNC_LDS 1
#else
#define HAVE_ASYNC_LDS 0
#endif

__device__ __forceinline__ void stage16_to_lds(const void* g, void* l) {
#if HAVE_ASYNC_LDS
    __builtin_amdgcn_global_load_async_to_lds_b128(
        (__attribute__((address_space(1))) i4v*)g,
        (__attribute__((address_space(3))) i4v*)l, 0, 0);
#else
    *(uint4*)l = *(const uint4*)g;
#endif
}

__device__ __forceinline__ void wait_stage_done() {
#if HAVE_ASYNC_LDS
#if __has_builtin(__builtin_amdgcn_s_wait_asynccnt)
    __builtin_amdgcn_s_wait_asynccnt(0);
#else
    asm volatile("s_wait_asynccnt 0x0" ::: "memory");
#endif
#endif
    __syncthreads();
}

// ---------------------------------------------------------------------------
// monotone-uint encoding of float for atomicMax-based segment max
// ---------------------------------------------------------------------------
__device__ __forceinline__ unsigned enc_f(float f) {
    unsigned u = __float_as_uint(f);
    return (u & 0x80000000u) ? ~u : (u | 0x80000000u);
}
__device__ __forceinline__ float dec_f(unsigned u) {
    return __uint_as_float((u & 0x80000000u) ? (u & 0x7fffffffu) : ~u);
}

// ---------------------------------------------------------------------------
// Embedding: h = features @ emb_w + emb_b  (K=12: VALU). Also emits bf16 copy.
// ---------------------------------------------------------------------------
__global__ __launch_bounds__(256)
void embed_kernel(const float* __restrict__ feat, const float* __restrict__ W,
                  const float* __restrict__ b, float* __restrict__ h,
                  __bf16* __restrict__ hbf, int n) {
    int t = blockIdx.x * 256 + threadIdx.x;
    if (t >= n * HID) return;
    int node = t >> 7, col = t & (HID - 1);
    float acc = b[col];
    const float* fp = feat + (size_t)node * IN_DIM;
#pragma unroll
    for (int k = 0; k < IN_DIM; ++k)
        acc = fmaf(fp[k], W[k * HID + col], acc);
    h[t]   = acc;
    hbf[t] = (__bf16)acc;
}

// ---------------------------------------------------------------------------
// W (K x M f32, row-major) -> WT (M x K bf16, row-major) ; tiny, once/layer
// ---------------------------------------------------------------------------
__global__ __launch_bounds__(256)
void transpose_w_bf16(const float* __restrict__ W, __bf16* __restrict__ WT,
                      int K, int M) {
    int t = blockIdx.x * 256 + threadIdx.x;
    if (t >= K * M) return;
    int m = t / K, k = t - m * K;
    WT[t] = (__bf16)W[(size_t)k * M + m];
}

// ---------------------------------------------------------------------------
// WMMA bf16 GEMM: C[nrows x M] = Abf[nrows x K](bf16) @ WT[M x K](bf16)^T
//
// block = 256 threads (8 waves); block tile = 128 rows x 32 cols; each wave
// computes 16 rows x 32 cols = two 16x16x32 WMMAs per k-step (A-frag reused).
//
// ISA 7.12.2 16-bit fragment layout is contiguous per lane:
//   lane (h,m)=16h+m holds K = 8h..8h+7 (elems 0..7) and 16+8h..16+8h+7
// so with bf16 row-major A (and row-major WT for B) a fragment is exactly
// two 16B loads per lane: A straight from global (rows are wave-private, no
// reuse -> no LDS), B from an 8KB LDS slab staged ONCE per block via
// GLOBAL_LOAD_ASYNC_TO_LDS_B128. No barriers inside the k-loop.
// Row guard = address clamp only (garbage rows are never stored).
// ---------------------------------------------------------------------------
template <int K, int M, bool BIASED, bool RELU>
__global__ __launch_bounds__(256)
void gemm_bf16_wmma(const __bf16* __restrict__ A, const __bf16* __restrict__ WT,
                    const float* __restrict__ bias, float* __restrict__ C,
                    int nrows) {
    __shared__ __align__(16) __bf16 Bt[32][K];  // 32 cols x K, 8KB @ K=128

    const int tid  = threadIdx.x;
    const int wave = tid >> 5;    // 0..7 -> 16-row sub-tile
    const int lane = tid & 31;
    const int rowBlock = blockIdx.x * 128;
    const int colBlock = blockIdx.y * 32;
    const int half = lane >> 4;
    const int m16  = lane & 15;

    // ---- stage B slab: rows [colBlock, colBlock+32) of WT = 32*K bf16,
    // contiguous in memory. 32*K*2/16 transfers, 256 threads.
    {
        const __bf16* gsrc = WT + (size_t)colBlock * K;
        constexpr int NXFER = (32 * K * 2) / 16;     // 512 @ K=128
#pragma unroll
        for (int i = tid; i < NXFER; i += 256)
            stage16_to_lds((const char*)gsrc + i * 16, (char*)&Bt[0][0] + i * 16);
    }
    wait_stage_done();

    // ---- per-lane fragment base pointers (all contiguous 16B loads)
    const int rowA  = rowBlock + wave * 16 + m16;
    const int rowAc = (rowA < nrows) ? rowA : (nrows - 1);   // clamp, no select
    const __bf16* arow = A + (size_t)rowAc * K + 8 * half;
    const __bf16* b0row = &Bt[m16][8 * half];        // col group 0
    const __bf16* b1row = &Bt[16 + m16][8 * half];   // col group 1

    v8f acc0 = {}, acc1 = {};
#pragma unroll
    for (int kt = 0; kt < K / 32; ++kt) {
        const int k0 = kt * 32;
        v8bf alo = *(const v8bf*)(arow + k0);
        v8bf ahi = *(const v8bf*)(arow + k0 + 16);
        v8bf blo0 = *(const v8bf*)(b0row + k0);
        v8bf bhi0 = *(const v8bf*)(b0row + k0 + 16);
        v8bf blo1 = *(const v8bf*)(b1row + k0);
        v8bf bhi1 = *(const v8bf*)(b1row + k0 + 16);
        v16bf afrag = __builtin_shufflevector(alo, ahi, 0,1,2,3,4,5,6,7,
                                              8,9,10,11,12,13,14,15);
        v16bf bf0   = __builtin_shufflevector(blo0, bhi0, 0,1,2,3,4,5,6,7,
                                              8,9,10,11,12,13,14,15);
        v16bf bf1   = __builtin_shufflevector(blo1, bhi1, 0,1,2,3,4,5,6,7,
                                              8,9,10,11,12,13,14,15);
        acc0 = __builtin_amdgcn_wmma_f32_16x16x32_bf16(false, afrag, false, bf0,
                                                       (short)0, acc0, false, false);
        acc1 = __builtin_amdgcn_wmma_f32_16x16x32_bf16(false, afrag, false, bf1,
                                                       (short)0, acc1, false, false);
    }

    // ---- epilogue: VGPR r -> row = 16*wave + 8*half + r, col = m16
    const int col0 = colBlock + m16;
    const int col1 = col0 + 16;
    const float bv0 = BIASED ? bias[col0] : 0.0f;
    const float bv1 = BIASED ? bias[col1] : 0.0f;
#pragma unroll
    for (int r = 0; r < 8; ++r) {
        int grow = rowBlock + wave * 16 + 8 * half + r;
        if (grow < nrows) {
            float v0 = acc0[r] + bv0;
            float v1 = acc1[r] + bv1;
            if (RELU) { v0 = fmaxf(v0, 0.0f); v1 = fmaxf(v1, 0.0f); }
            C[(size_t)grow * M + col0] = v0;
            C[(size_t)grow * M + col1] = v1;
        }
    }
}

// ---------------------------------------------------------------------------
// el/er = per-head dot(z, attn_{l,r});  one thread per (node, head)
// ---------------------------------------------------------------------------
__global__ __launch_bounds__(256)
void attn_proj_kernel(const float* __restrict__ z, const float* __restrict__ al,
                      const float* __restrict__ ar, float* __restrict__ el,
                      float* __restrict__ er, int n) {
    int t = blockIdx.x * 256 + threadIdx.x;
    if (t >= n * HEADS) return;
    int node = t >> 2, hd = t & (HEADS - 1);
    const float* zp  = z + (size_t)node * HID + hd * DHEAD;
    const float* alp = al + hd * DHEAD;
    const float* arp = ar + hd * DHEAD;
    float a = 0.f, b = 0.f;
#pragma unroll
    for (int d = 0; d < DHEAD; ++d) {
        float zv = zp[d];
        a = fmaf(zv, alp[d], a);
        b = fmaf(zv, arp[d], b);
    }
    el[t] = a;
    er[t] = b;
}

// ---------------------------------------------------------------------------
// per-layer init: m = enc(-inf), denom = 0, rst = 0
// ---------------------------------------------------------------------------
__global__ __launch_bounds__(256)
void init_layer_kernel(unsigned* __restrict__ menc, float* __restrict__ denom,
                       float* __restrict__ rst, int n) {
    int t = blockIdx.x * 256 + threadIdx.x;
    if (t < n * HEADS) { menc[t] = 0x007FFFFFu; denom[t] = 0.0f; }  // enc(-inf)
    if (t < n * HID) rst[t] = 0.0f;
}

__device__ __forceinline__ float edge_e(const float* el, const float* er,
                                        int s, int d, int hd) {
    float x = el[s * HEADS + hd] + er[d * HEADS + hd];
    return (x > 0.0f) ? x : NEG_SLOPE * x;
}

// pass 1: segment max over dst (atomic max on monotone-uint encoding)
__global__ __launch_bounds__(256)
void edge_max_kernel(const int* __restrict__ src, const int* __restrict__ dst,
                     const float* __restrict__ el, const float* __restrict__ er,
                     unsigned* __restrict__ menc, int E) {
    int t = blockIdx.x * 256 + threadIdx.x;
    if (t >= E * HEADS) return;
    int e = t >> 2, hd = t & (HEADS - 1);
    int s = src[e], d = dst[e];
    float x = edge_e(el, er, s, d, hd);
    atomicMax(&menc[d * HEADS + hd], enc_f(x));
}

// pass 2: denom = segment_sum(exp(e - m))
__global__ __launch_bounds__(256)
void edge_denom_kernel(const int* __restrict__ src, const int* __restrict__ dst,
                       const float* __restrict__ el, const float* __restrict__ er,
                       const unsigned* __restrict__ menc, float* __restrict__ denom,
                       int E) {
    int t = blockIdx.x * 256 + threadIdx.x;
    if (t >= E * HEADS) return;
    int e = t >> 2, hd = t & (HEADS - 1);
    int s = src[e], d = dst[e];
    float x = edge_e(el, er, s, d, hd);
    float mv = dec_f(menc[d * HEADS + hd]);
    if (!(fabsf(mv) < 3.0e38f)) mv = 0.0f;  // guard empty/inf segments
    atomicAdd(&denom[d * HEADS + hd], __expf(x - mv));
}

// pass 3: rst[dst] += alpha * z[src]; one thread per (edge, channel)
__global__ __launch_bounds__(256)
void edge_aggr_kernel(const int* __restrict__ src, const int* __restrict__ dst,
                      const float* __restrict__ el, const float* __restrict__ er,
                      const unsigned* __restrict__ menc, const float* __restrict__ denom,
                      const float* __restrict__ z, float* __restrict__ rst,
                      long long total) {
    long long t = (long long)blockIdx.x * 256 + threadIdx.x;
    if (t >= total) return;
    int e   = (int)(t >> 7);
    int dim = (int)(t & (HID - 1));
    int hd  = dim >> 5;
    int s = src[e], d = dst[e];
    float x = edge_e(el, er, s, d, hd);
    float mv = dec_f(menc[d * HEADS + hd]);
    if (!(fabsf(mv) < 3.0e38f)) mv = 0.0f;
    float alpha = __expf(x - mv) / fmaxf(denom[d * HEADS + hd], 1e-20f);
    atomicAdd(&rst[(size_t)d * HID + dim], z[(size_t)s * HID + dim] * alpha);
}

// residual + bias + BN(eval) + ReLU, in place into h; also emits bf16 copy
__global__ __launch_bounds__(256)
void finalize_kernel(float* __restrict__ h, __bf16* __restrict__ hbf,
                     const float* __restrict__ rst,
                     const float* __restrict__ bias, const float* __restrict__ g,
                     const float* __restrict__ b, const float* __restrict__ rm,
                     const float* __restrict__ rv, int n) {
    int t = blockIdx.x * 256 + threadIdx.x;
    if (t >= n * HID) return;
    int col = t & (HID - 1);
    float v = rst[t] + h[t] + bias[col];
    v = (v - rm[col]) * rsqrtf(rv[col] + BN_EPS) * g[col] + b[col];
    v = fmaxf(v, 0.0f);
    h[t]   = v;
    hbf[t] = (__bf16)v;
}

// fc2: (N,64) @ (64,2) + b ; one thread per node
__global__ __launch_bounds__(256)
void fc2_kernel(const float* __restrict__ x, const float* __restrict__ W,
                const float* __restrict__ b, float* __restrict__ out, int n) {
    int node = blockIdx.x * 256 + threadIdx.x;
    if (node >= n) return;
    const float* xp = x + (size_t)node * 64;
    float a0 = b[0], a1 = b[1];
#pragma unroll
    for (int k = 0; k < 64; ++k) {
        float xv = xp[k];
        a0 = fmaf(xv, W[k * 2 + 0], a0);
        a1 = fmaf(xv, W[k * 2 + 1], a1);
    }
    out[node * 2 + 0] = a0;
    out[node * 2 + 1] = a1;
}

// ---------------------------------------------------------------------------
// host launcher
// ---------------------------------------------------------------------------
extern "C" void kernel_launch(void* const* d_in, const int* in_sizes, int n_in,
                              void* d_out, int out_size, void* d_ws, size_t ws_size,
                              hipStream_t stream) {
    const int N = in_sizes[0] / IN_DIM;
    const int E = in_sizes[1] / 2;

    // inputs in setup_inputs() insertion order
    const float* feat = (const float*)d_in[0];
    const int*   eidx = (const int*)d_in[1];
    const int*   src  = eidx;
    const int*   dst  = eidx + E;
    const float* emb_w = (const float*)d_in[2];
    const float* emb_b = (const float*)d_in[3];
    // layers: 8 tensors each: W, attn_l, attn_r, bias, bn_g, bn_b, bn_rm, bn_rv
    const float* Lp[3][8];
    for (int l = 0; l < 3; ++l)
        for (int j = 0; j < 8; ++j)
            Lp[l][j] = (const float*)d_in[4 + l * 8 + j];
    const float* fc1_w = (const float*)d_in[28];
    const float* fc1_b = (const float*)d_in[29];
    const float* fc2_w = (const float*)d_in[30];
    const float* fc2_b = (const float*)d_in[31];
    float* out = (float*)d_out;

    // workspace carve-out (256B aligned)
    size_t off = 0;
    auto carve = [&](size_t bytes) -> char* {
        char* p = (char*)d_ws + off;
        off += (bytes + 255) & ~(size_t)255;
        return p;
    };
    float*    h     = (float*)carve((size_t)N * HID * 4);
    float*    z     = (float*)carve((size_t)N * HID * 4);   // also reused as fc1 out
    float*    rst   = (float*)carve((size_t)N * HID * 4);
    __bf16*   hbf   = (__bf16*)carve((size_t)N * HID * 2);
    __bf16*   wtbf  = (__bf16*)carve((size_t)HID * HID * 2); // W^T / fc1_w^T
    float*    el    = (float*)carve((size_t)N * HEADS * 4);
    float*    er    = (float*)carve((size_t)N * HEADS * 4);
    unsigned* menc  = (unsigned*)carve((size_t)N * HEADS * 4);
    float*    denom = (float*)carve((size_t)N * HEADS * 4);
    (void)ws_size; (void)n_in; (void)out_size;

    const int nh  = N * HID;
    const int bNH = (nh + 255) / 256;
    const int bN4 = (N * HEADS + 255) / 256;
    const int bE4 = (E * HEADS + 255) / 256;
    const long long aggr_total = (long long)E * HID;
    const unsigned bAggr = (unsigned)((aggr_total + 255) / 256);
    const dim3 gemmGridZ((N + 127) / 128, HID / 32);
    const dim3 gemmGridF((N + 127) / 128, 64 / 32);

    embed_kernel<<<bNH, 256, 0, stream>>>(feat, emb_w, emb_b, h, hbf, N);

    for (int l = 0; l < 3; ++l) {
        const float* W      = Lp[l][0];
        const float* attn_l = Lp[l][1];
        const float* attn_r = Lp[l][2];
        const float* bias   = Lp[l][3];
        const float* bn_g   = Lp[l][4];
        const float* bn_b   = Lp[l][5];
        const float* bn_rm  = Lp[l][6];
        const float* bn_rv  = Lp[l][7];

        transpose_w_bf16<<<(HID * HID + 255) / 256, 256, 0, stream>>>(W, wtbf,
                                                                      HID, HID);
        gemm_bf16_wmma<HID, HID, false, false>
            <<<gemmGridZ, 256, 0, stream>>>(hbf, wtbf, nullptr, z, N);
        attn_proj_kernel<<<bN4, 256, 0, stream>>>(z, attn_l, attn_r, el, er, N);
        init_layer_kernel<<<bNH, 256, 0, stream>>>(menc, denom, rst, N);
        edge_max_kernel<<<bE4, 256, 0, stream>>>(src, dst, el, er, menc, E);
        edge_denom_kernel<<<bE4, 256, 0, stream>>>(src, dst, el, er, menc, denom, E);
        edge_aggr_kernel<<<bAggr, 256, 0, stream>>>(src, dst, el, er, menc, denom,
                                                    z, rst, aggr_total);
        finalize_kernel<<<bNH, 256, 0, stream>>>(h, hbf, rst, bias, bn_g, bn_b,
                                                 bn_rm, bn_rv, N);
    }

    // fc1 (WMMA, bias+ReLU epilogue) into z buffer, then fc2
    transpose_w_bf16<<<(HID * 64 + 255) / 256, 256, 0, stream>>>(fc1_w, wtbf,
                                                                 HID, 64);
    gemm_bf16_wmma<HID, 64, true, true>
        <<<gemmGridF, 256, 0, stream>>>(hbf, wtbf, fc1_b, z, N);
    fc2_kernel<<<(N + 255) / 256, 256, 0, stream>>>(z, fc2_w, fc2_b, out, N);
}
